// GraphATABase_58712202936398
// MI455X (gfx1250) — compile-verified, matrix operand order checked by reading, not run on hardware
//
#include <hip/hip_runtime.h>

#define HIDD   64
#define MMOD   3
#define ATTH   16
#define COUT   10

typedef __attribute__((ext_vector_type(16))) __bf16 v16bf;
typedef __attribute__((ext_vector_type(8)))  __bf16 v8bf;
typedef __attribute__((ext_vector_type(4)))  __bf16 v4bf;
typedef __attribute__((ext_vector_type(8)))  float  v8f;

// K index within a 16x32 bf16 A/B WMMA fragment for element j (0..15) of `lane`
// (CDNA5 ISA 7.12.2: VGPR0..3 hold K={0..7}+8*lanegroup, VGPR4..7 hold K=16+{0..7}+8*lanegroup)
static __device__ __forceinline__ int frag_k(int lane, int j) {
  int v = j >> 1, pos = j & 1, g = lane >> 4;
  return (v < 4) ? (g * 8 + v * 2 + pos) : (16 + g * 8 + (v - 4) * 2 + pos);
}

static __device__ __forceinline__ v16bf combine8(v8bf lo, v8bf hi) {
  return __builtin_shufflevector(lo, hi, 0, 1, 2, 3, 4, 5, 6, 7, 8, 9, 10, 11, 12, 13, 14, 15);
}

// ---------------------------------------------------------------------------
// One-shot prep kernels
// ---------------------------------------------------------------------------

// Pack a (Mp, IN, OUT) f32 weight tensor into bf16 WMMA-B fragment order:
// frag (m,kt,nt) -> 32 lanes x 16 contiguous bf16.  One thread per (frag,lane).
__global__ void pack_w_kernel(const float* __restrict__ W, __bf16* __restrict__ Wp,
                              int Mp, int IN, int OUT) {
  int KT = IN / 32, NT = OUT / 16;
  int total = Mp * KT * NT * 32;
  int t = blockIdx.x * blockDim.x + threadIdx.x;
  if (t >= total) return;
  int lane = t & 31;
  int f = t >> 5;                 // (m*KT + kt)*NT + nt
  int nt = f % NT;
  int kt = (f / NT) % KT;
  int m  = f / (NT * KT);
  int col = lane & 15;
  __bf16* out = Wp + (size_t)t * 16;
#pragma unroll
  for (int j = 0; j < 16; ++j) {
    int K = kt * 32 + frag_k(lane, j);
    out[j] = (__bf16)W[((size_t)m * IN + K) * OUT + nt * 16 + col];
  }
}

// f32 -> bf16 (4 elements per thread), optional fused ReLU
template <bool RELU>
__global__ void cvt_bf16_kernel(const float* __restrict__ in, __bf16* __restrict__ out,
                                int total4) {
  int i = blockIdx.x * blockDim.x + threadIdx.x;
  if (i >= total4) return;
  float4 v = ((const float4*)in)[i];
  if (RELU) {
    v.x = fmaxf(v.x, 0.f); v.y = fmaxf(v.y, 0.f);
    v.z = fmaxf(v.z, 0.f); v.w = fmaxf(v.w, 0.f);
  }
  v4bf o;
  o[0] = (__bf16)v.x; o[1] = (__bf16)v.y; o[2] = (__bf16)v.z; o[3] = (__bf16)v.w;
  ((v4bf*)out)[i] = o;
}

// ---------------------------------------------------------------------------
// Fused: z_m = X @ W[m] (WMMA), s_m = tanh(z_m@aw1+b)@aw2 (WMMA + shfl reduce),
// h = sum_m softmax(s)_m * z_m  (directly from live accumulators).
// One wave = one 16-node tile; 4 waves / 128-thread block.
// ---------------------------------------------------------------------------
template <int INDIM>
__global__ void __launch_bounds__(128)
fused_gemm_att_kernel(const __bf16* __restrict__ Xb,   // (N, INDIM) bf16 (relu pre-applied)
                      const __bf16* __restrict__ Wp,   // packed W frags (M,KT,NT,32,16)
                      const __bf16* __restrict__ AWp,  // packed att_w1 frags (2,32,16), K=64
                      const float* __restrict__ ab1,   // (ATTH)
                      const float* __restrict__ aw2,   // (ATTH)
                      float* __restrict__ Hout,        // (N, HIDD)
                      int nNodes) {
  constexpr int WAVES = 4;
  constexpr int KT    = INDIM / 32;
  constexpr int NT    = HIDD / 16;       // 4
  constexpr int ZROW  = HIDD + 8;        // padded bf16 row (spread LDS banks)
  __shared__ __align__(16) __bf16 zb[WAVES][MMOD][16][ZROW];
  __shared__ float sbuf[WAVES][16][MMOD];
  __shared__ float bbuf[WAVES][16][MMOD];

  const int lane  = threadIdx.x & 31;
  const int wid   = threadIdx.x >> 5;
  const int node0 = (blockIdx.x * WAVES + wid) * 16;
  const int nrow  = nNodes - node0;      // valid rows (may be <= 0)
  const int row   = lane & 15;           // A-row / B-col / C-col owned by this lane
  const int g     = lane >> 4;

  // ---- A fragments: two aligned 16B bf16 runs per lane per K-step ----
  const bool rowok = (row < nrow);
  const __bf16* xrow = Xb + (size_t)(rowok ? (node0 + row) : 0) * INDIM;
  v16bf afrag[KT];
#pragma unroll
  for (int kt = 0; kt < KT; ++kt)
    afrag[kt] = combine8(*(const v8bf*)(xrow + kt * 32 + g * 8),
                         *(const v8bf*)(xrow + kt * 32 + 16 + g * 8));

  // ---- main GEMM: 3 models x 4 column tiles, accumulators stay live ----
  v8f acc[MMOD][NT];
#pragma unroll
  for (int m = 0; m < MMOD; ++m)
#pragma unroll
    for (int nt = 0; nt < NT; ++nt) {
      v8f a = {};
#pragma unroll
      for (int kt = 0; kt < KT; ++kt) {
        const __bf16* wp = Wp + (size_t)((((m * KT + kt) * NT) + nt) * 32 + lane) * 16;
        v16bf bfrag = combine8(*(const v8bf*)wp, *(const v8bf*)(wp + 8));
        a = __builtin_amdgcn_wmma_f32_16x16x32_bf16(false, afrag[kt], false, bfrag,
                                                    (short)0, a, false, false);
      }
      acc[m][nt] = a;
      // spill z as bf16 for the attention matmul (C layout: row=r+8g, col=lane&15)
#pragma unroll
      for (int r = 0; r < 8; ++r)
        zb[wid][m][r + 8 * g][nt * 16 + row] = (__bf16)a[r];
    }
  __syncthreads();

  // ---- attention scores via WMMA: t = z_m @ aw1  (16x16, K=64 -> 2 steps) ----
  const float bias = ab1[row];   // attention column j owned by this lane
  const float w2j  = aw2[row];
#pragma unroll
  for (int m = 0; m < MMOD; ++m) {
    v8f t = {};
#pragma unroll
    for (int kt = 0; kt < 2; ++kt) {
      const __bf16* zr = &zb[wid][m][row][0];
      v16bf az = combine8(*(const v8bf*)(zr + kt * 32 + g * 8),
                          *(const v8bf*)(zr + kt * 32 + 16 + g * 8));
      const __bf16* ap = AWp + (size_t)(kt * 32 + lane) * 16;
      v16bf bz = combine8(*(const v8bf*)ap, *(const v8bf*)(ap + 8));
      t = __builtin_amdgcn_wmma_f32_16x16x32_bf16(false, az, false, bz,
                                                  (short)0, t, false, false);
    }
    float p[8];
#pragma unroll
    for (int r = 0; r < 8; ++r) p[r] = tanhf(t[r] + bias) * w2j;
    // sum over the 16 attention columns: xor-reduce within each 16-lane half
#pragma unroll
    for (int off = 1; off < 16; off <<= 1)
#pragma unroll
      for (int r = 0; r < 8; ++r) p[r] += __shfl_xor(p[r], off, 32);
    if (row == 0)
#pragma unroll
      for (int r = 0; r < 8; ++r) sbuf[wid][r + 8 * g][m] = p[r];
  }
  __syncthreads();

  // ---- softmax over M per node ----
  if (lane < 16) {
    float s0 = sbuf[wid][lane][0], s1 = sbuf[wid][lane][1], s2 = sbuf[wid][lane][2];
    float mx = fmaxf(s0, fmaxf(s1, s2));
    float e0 = __expf(s0 - mx), e1 = __expf(s1 - mx), e2 = __expf(s2 - mx);
    float inv = 1.f / (e0 + e1 + e2);
    bbuf[wid][lane][0] = e0 * inv;
    bbuf[wid][lane][1] = e1 * inv;
    bbuf[wid][lane][2] = e2 * inv;
  }
  __syncthreads();

  // ---- beta-weighted sum straight from accumulators, coalesced stores ----
#pragma unroll
  for (int r = 0; r < 8; ++r) {
    int nloc = r + 8 * g;
    if (nloc < nrow) {
      float b0 = bbuf[wid][nloc][0], b1 = bbuf[wid][nloc][1], b2 = bbuf[wid][nloc][2];
      float* out = Hout + (size_t)(node0 + nloc) * HIDD;
#pragma unroll
      for (int nt = 0; nt < NT; ++nt)
        out[nt * 16 + row] =
            b0 * acc[0][nt][r] + b1 * acc[1][nt][r] + b2 * acc[2][nt][r];
    }
  }
}

// ---------------------------------------------------------------------------
// Degree / normalization / propagation kernels
// ---------------------------------------------------------------------------
__global__ void zero_f32(float* __restrict__ p, int n) {
  int i = blockIdx.x * blockDim.x + threadIdx.x;
  if (i < n) p[i] = 0.f;
}

__global__ void deg_kernel(const int* __restrict__ dst, float* __restrict__ deg, int e) {
  int i = blockIdx.x * blockDim.x + threadIdx.x;
  if (i < e) atomicAdd(&deg[dst[i]], 1.0f);
}

__global__ void dis_kernel(const float* __restrict__ deg, float* __restrict__ dis, int n) {
  int i = blockIdx.x * blockDim.x + threadIdx.x;
  if (i < n) dis[i] = rsqrtf(deg[i] + 1.0f);
}

// agg = dis^2 * h_pre  (self-loop term; initializes agg buffer)
__global__ void selfterm_kernel(const float* __restrict__ hpre, const float* __restrict__ dis,
                                float* __restrict__ agg, int total) {
  int i = blockIdx.x * blockDim.x + threadIdx.x;
  if (i < total) {
    int node = i / HIDD;
    float s = dis[node];
    agg[i] = s * s * hpre[i];
  }
}

// 16 lanes per edge: float4 gather of h_pre[src], scaled, atomic scatter to agg[dst]
__global__ void scatter_kernel(const int* __restrict__ src, const int* __restrict__ dst,
                               const float* __restrict__ dis, const float* __restrict__ hpre,
                               float* __restrict__ agg, int e) {
  long long t = (long long)blockIdx.x * blockDim.x + threadIdx.x;
  int eidx = (int)(t >> 4);
  int r    = (int)(t & 15);
  if (eidx >= e) return;
  __builtin_prefetch(src + eidx + 4096, 0, 1);   // global_prefetch_b8 on edge stream
  __builtin_prefetch(dst + eidx + 4096, 0, 1);
  int s = src[eidx], d = dst[eidx];
  float norm = dis[s] * dis[d];
  const float4 v = *(const float4*)(hpre + (size_t)s * HIDD + r * 4);
  float* a = agg + (size_t)d * HIDD + r * 4;
  atomicAdd(a + 0, v.x * norm);
  atomicAdd(a + 1, v.y * norm);
  atomicAdd(a + 2, v.z * norm);
  atomicAdd(a + 3, v.w * norm);
}

// ---------------------------------------------------------------------------
// Classifier ensemble + class attention + log_softmax (thread per node)
// ---------------------------------------------------------------------------
__global__ void cls_kernel(const float* __restrict__ h,     // (N, HIDD)
                           const float* __restrict__ clsW,  // (M, HIDD, COUT)
                           const float* __restrict__ clsb,  // (M, COUT)
                           const float* __restrict__ aw1,   // (COUT, ATTH)
                           const float* __restrict__ ab1,   // (ATTH)
                           const float* __restrict__ aw2,   // (ATTH)
                           float* __restrict__ out, int n) {
  int i = blockIdx.x * blockDim.x + threadIdx.x;
  if (i >= n) return;
  float hv[HIDD];
  const float4* hp = (const float4*)(h + (size_t)i * HIDD);
#pragma unroll
  for (int q = 0; q < HIDD / 4; ++q) {
    float4 v = hp[q];
    hv[q * 4 + 0] = v.x; hv[q * 4 + 1] = v.y; hv[q * 4 + 2] = v.z; hv[q * 4 + 3] = v.w;
  }
  float logits[MMOD][COUT];
#pragma unroll
  for (int m = 0; m < MMOD; ++m)
#pragma unroll
    for (int c = 0; c < COUT; ++c) {
      float acc = clsb[m * COUT + c];
#pragma unroll 8
      for (int d = 0; d < HIDD; ++d) acc += hv[d] * clsW[((size_t)m * HIDD + d) * COUT + c];
      logits[m][c] = acc;
    }
  float s[MMOD];
#pragma unroll
  for (int m = 0; m < MMOD; ++m) {
    float sm = 0.f;
#pragma unroll
    for (int j = 0; j < ATTH; ++j) {
      float t = ab1[j];
#pragma unroll
      for (int c = 0; c < COUT; ++c) t += logits[m][c] * aw1[c * ATTH + j];
      sm += tanhf(t) * aw2[j];
    }
    s[m] = sm;
  }
  float mx = fmaxf(s[0], fmaxf(s[1], s[2]));
  float e0 = __expf(s[0] - mx), e1 = __expf(s[1] - mx), e2 = __expf(s[2] - mx);
  float inv = 1.f / (e0 + e1 + e2);
  float b0 = e0 * inv, b1 = e1 * inv, b2 = e2 * inv;
  float o[COUT];
  float omax = -1e30f;
#pragma unroll
  for (int c = 0; c < COUT; ++c) {
    o[c] = b0 * logits[0][c] + b1 * logits[1][c] + b2 * logits[2][c];
    omax = fmaxf(omax, o[c]);
  }
  float sum = 0.f;
#pragma unroll
  for (int c = 0; c < COUT; ++c) sum += __expf(o[c] - omax);
  float lse = omax + __logf(sum);
  float* op = out + (size_t)i * COUT;
#pragma unroll
  for (int c = 0; c < COUT; ++c) op[c] = o[c] - lse;
}

// ---------------------------------------------------------------------------
extern "C" void kernel_launch(void* const* d_in, const int* in_sizes, int n_in,
                              void* d_out, int out_size, void* d_ws, size_t ws_size,
                              hipStream_t stream) {
  const float* x    = (const float*)d_in[0];
  const int*   ei   = (const int*)d_in[1];
  const float* W1   = (const float*)d_in[2];
  const float* a1w1 = (const float*)d_in[3];
  const float* a1b1 = (const float*)d_in[4];
  const float* a1w2 = (const float*)d_in[5];
  const float* W2   = (const float*)d_in[6];
  const float* a2w1 = (const float*)d_in[7];
  const float* a2b1 = (const float*)d_in[8];
  const float* a2w2 = (const float*)d_in[9];
  const float* clsW = (const float*)d_in[10];
  const float* clsb = (const float*)d_in[11];
  const float* acw1 = (const float*)d_in[12];
  const float* acb1 = (const float*)d_in[13];
  const float* acw2 = (const float*)d_in[14];

  const int INDIM = 128;
  const int n = in_sizes[0] / INDIM;
  const int e = in_sizes[1] / 2;
  const int* src = ei;
  const int* dst = ei + e;

  // workspace carve (~90 MB total)
  char* ws = (char*)d_ws;
  auto carve = [&](size_t bytes) -> char* {
    char* p = ws;
    ws += (bytes + 255) & ~(size_t)255;
    return p;
  };
  float*  deg  = (float*)carve((size_t)n * 4);
  float*  dis  = (float*)carve((size_t)n * 4);
  float*  bufA = (float*)carve((size_t)n * HIDD * 4);
  float*  bufB = (float*)carve((size_t)n * HIDD * 4);
  __bf16* Xb   = (__bf16*)carve((size_t)n * INDIM * 2);
  __bf16* Hb   = (__bf16*)carve((size_t)n * HIDD * 2);
  __bf16* Wp1  = (__bf16*)carve((size_t)MMOD * (INDIM / 32) * (HIDD / 16) * 512 * 2);
  __bf16* Wp2  = (__bf16*)carve((size_t)MMOD * (HIDD / 32) * (HIDD / 16) * 512 * 2);
  __bf16* AWp1 = (__bf16*)carve((size_t)(HIDD / 32) * 512 * 2);
  __bf16* AWp2 = (__bf16*)carve((size_t)(HIDD / 32) * 512 * 2);

  const int B = 256;

  // one-shot weight packing (bf16, WMMA fragment order)
  pack_w_kernel<<<(MMOD * (INDIM / 32) * (HIDD / 16) * 32 + B - 1) / B, B, 0, stream>>>(
      W1, Wp1, MMOD, INDIM, HIDD);
  pack_w_kernel<<<(MMOD * (HIDD / 32) * (HIDD / 16) * 32 + B - 1) / B, B, 0, stream>>>(
      W2, Wp2, MMOD, HIDD, HIDD);
  pack_w_kernel<<<1, B, 0, stream>>>(a1w1, AWp1, 1, HIDD, ATTH);
  pack_w_kernel<<<1, B, 0, stream>>>(a2w1, AWp2, 1, HIDD, ATTH);

  // activations -> bf16
  cvt_bf16_kernel<false><<<((n * INDIM / 4) + B - 1) / B, B, 0, stream>>>(x, Xb, n * INDIM / 4);

  // degrees + D^-1/2
  zero_f32<<<(n + B - 1) / B, B, 0, stream>>>(deg, n);
  deg_kernel<<<(e + B - 1) / B, B, 0, stream>>>(dst, deg, e);
  dis_kernel<<<(n + B - 1) / B, B, 0, stream>>>(deg, dis, n);

  const int tiles   = (n + 15) / 16;
  const int gblocks = (tiles + 3) / 4;     // 4 wave-tiles per 128-thread block
  const int tot     = n * HIDD;
  const long long sthreads = (long long)e * 16;
  const unsigned sblocks = (unsigned)((sthreads + B - 1) / B);

  // layer 1: Xb @ W1 (+attention) -> bufA ; propagate -> bufB
  fused_gemm_att_kernel<128><<<gblocks, 128, 0, stream>>>(Xb, Wp1, AWp1, a1b1, a1w2, bufA, n);
  selfterm_kernel<<<(tot + B - 1) / B, B, 0, stream>>>(bufA, dis, bufB, tot);
  scatter_kernel<<<sblocks, B, 0, stream>>>(src, dst, dis, bufA, bufB, e);

  // layer 2: relu(bufB) -> bf16 Hb ; Hb @ W2 (+attention) -> bufA ; propagate -> bufB
  cvt_bf16_kernel<true><<<((tot / 4) + B - 1) / B, B, 0, stream>>>(bufB, Hb, tot / 4);
  fused_gemm_att_kernel<64><<<gblocks, 128, 0, stream>>>(Hb, Wp2, AWp2, a2b1, a2w2, bufA, n);
  selfterm_kernel<<<(tot + B - 1) / B, B, 0, stream>>>(bufA, dis, bufB, tot);
  scatter_kernel<<<sblocks, B, 0, stream>>>(src, dst, dis, bufA, bufB, e);

  // classifier ensemble + attention + log_softmax -> d_out (N, 10)
  cls_kernel<<<(n + B - 1) / B, B, 0, stream>>>(bufB, clsW, clsb, acw1, acb1, acw2,
                                                (float*)d_out, n);
}
